// MultiHeadAttention_30734785970264
// MI455X (gfx1250) — matrix operand order repaired and checked by reading, not compiled, not run on hardware
//
#include <hip/hip_runtime.h>
#include <hip/hip_bf16.h>

// Problem constants (from reference)
#define BATCH 4
#define SEQ   4096      // S, power of two (>>12, &4095)
#define NH    16        // heads
#define DH    64        // head dim
#define DDIM  1024      // D = NH*DH
#define MDIM  (BATCH*SEQ)   // 16384 rows
#define KDIM  1024          // contraction dim of every GEMM
#define KSTEPS (KDIM/32)
#define EPSV  1e-6f

typedef __attribute__((ext_vector_type(16))) __bf16 v16bf;
typedef __attribute__((ext_vector_type(8)))  float  v8f;

__device__ __forceinline__ float elup1(float x) {
    // elu(x)+1 : x>0 ? x+1 : exp(x)
    return x > 0.f ? x + 1.f : __expf(x);
}

__device__ __forceinline__ float4 scl4(float4 a, float s) {
    return make_float4(a.x * s, a.y * s, a.z * s, a.w * s);
}

__device__ __forceinline__ v16bf cvt16(float4 a0, float4 a1, float4 a2, float4 a3) {
    v16bf r;
    r[0]  = (__bf16)a0.x; r[1]  = (__bf16)a0.y; r[2]  = (__bf16)a0.z; r[3]  = (__bf16)a0.w;
    r[4]  = (__bf16)a1.x; r[5]  = (__bf16)a1.y; r[6]  = (__bf16)a1.z; r[7]  = (__bf16)a1.w;
    r[8]  = (__bf16)a2.x; r[9]  = (__bf16)a2.y; r[10] = (__bf16)a2.z; r[11] = (__bf16)a2.w;
    r[12] = (__bf16)a3.x; r[13] = (__bf16)a3.y; r[14] = (__bf16)a3.z; r[15] = (__bf16)a3.w;
    return r;
}

// --- CDNA5 async global->LDS copy (raw 16B per lane, tracked on ASYNCcnt) ---
__device__ __forceinline__ void async_cp16(uint32_t lds_off, const float* gsrc) {
    asm volatile("global_load_async_to_lds_b128 %0, %1, off"
                 :: "v"(lds_off), "v"(gsrc) : "memory");
}
__device__ __forceinline__ void async_cp16_off16(uint32_t lds_off, const float* gsrc) {
    // INST_OFFSET applies to both LDS and global addresses
    asm volatile("global_load_async_to_lds_b128 %0, %1, off offset:16"
                 :: "v"(lds_off), "v"(gsrc) : "memory");
}
__device__ __forceinline__ void wait_async(int n) {
#if __has_builtin(__builtin_amdgcn_s_wait_asynccnt)
    if (n == 0) __builtin_amdgcn_s_wait_asynccnt(0);
    else        __builtin_amdgcn_s_wait_asynccnt(2);
#else
    if (n == 0) asm volatile("s_wait_asynccnt 0x0" ::: "memory");
    else        asm volatile("s_wait_asynccnt 0x2" ::: "memory");
#endif
}

// ---------------------------------------------------------------------------
// Tiled WMMA GEMM:  C[m,n] = sum_k X[m,k] * W[n,k]   (X: MxK, W: NxK, both f32
// row-major -> torch Linear x @ W.T).  bf16 operands, f32 accumulate.
// Block = 256 threads = 8 waves; wave owns 16(M) x 64(N), block 128(M) x 64(N).
// B tile (64 rows x 32 K, f32) staged per-block into double-buffered LDS with
// global_load_async_to_lds_b128, consumed via ds_load_b128 + cvt to bf16.
//
// mode 0: D = elu(C)+1            -> out[m*1024 + n]              (qf)
// mode 1/2: D = C                 -> out[((b*H+h)*S+s)*64 + d]    (k/v raw)
// mode 3: A pre-scaled by c[b,h]*Z[b,h,s]; D = C -> out[m*1024+n] (hidden)
// ---------------------------------------------------------------------------
__global__ __launch_bounds__(256)
void gemm_wmma_bf16_kernel(const float* __restrict__ X,
                           const float* __restrict__ W,
                           float* __restrict__ out,
                           const float* __restrict__ cArr,   // [B*H]   (mode 3)
                           const float* __restrict__ Zarr,   // [B*H*S] (mode 3)
                           int mode) {
    __shared__ float ldsB[2][64][32];   // 2 x 8KB double buffer (f32 B tile)

    const int tid  = threadIdx.x;
    const int lane = tid & 31;
    const int wave = tid >> 5;
    const int hi   = lane >> 4;          // 0: lanes 0-15, 1: lanes 16-31
    const int lo   = lane & 15;

    const int m0 = blockIdx.x * 128 + wave * 16;   // wave's M base
    const int n0 = blockIdx.y * 64;                // block's N base
    const int mrow = m0 + lo;                      // A row held by this lane

    const float* __restrict__ xrow = X + (size_t)mrow * KDIM;
    const int bb = mrow >> 12;         // mrow / SEQ
    const int ss = mrow & (SEQ - 1);   // mrow % SEQ

    // staging geometry: 256 threads cover 64 rows x 32 K, 8 floats (32B) each
    const int srow = tid >> 2;
    const int sq8  = (tid & 3) * 8;
    const uint32_t lds0 = (uint32_t)(size_t)(void*)&ldsB[0][0][0];
    const uint32_t myLds = lds0 + (uint32_t)(srow * 32 + sq8) * 4u;  // buf 0 byte addr
    const float* __restrict__ wsrc = W + (size_t)(n0 + srow) * KDIM + sq8;

    v8f acc0 = {}, acc1 = {}, acc2 = {}, acc3 = {};

    // preload k-step 0 into buffer 0
    async_cp16_off16(myLds, wsrc);     // copies bytes [16,32) only

    for (int kk = 0; kk < KSTEPS; ++kk) {
        const int k   = kk * 32;
        const int cur = kk & 1;

        __syncthreads();               // everyone done reading buffer cur^1 (iter kk-1)
        if (kk + 1 < KSTEPS) {
            const uint32_t l = myLds + (uint32_t)((cur ^ 1) * 64 * 32 * 4);
            async_cp16_off16(l, wsrc + (kk + 1) * 32);
        }
        wait_async(kk + 1 < KSTEPS ? 2 : 0);   // this wave's copies for buf cur done
        __syncthreads();                       // tile globally visible

        // ---- A fragment: ISA 16-bit A 16x32 layout (direct from global).
        const int c0 = k + hi * 8;
        const int c1 = k + 16 + hi * 8;
        float4 a0 = *(const float4*)(xrow + c0);
        float4 a1 = *(const float4*)(xrow + c0 + 4);
        float4 a2 = *(const float4*)(xrow + c1);
        float4 a3 = *(const float4*)(xrow + c1 + 4);
        if (mode == 3) {
            const int h0 = c0 >> 6, h1 = c1 >> 6;
            const float s0 = cArr[bb * NH + h0] * Zarr[((size_t)(bb * NH + h0)) * SEQ + ss];
            const float s1 = cArr[bb * NH + h1] * Zarr[((size_t)(bb * NH + h1)) * SEQ + ss];
            a0 = scl4(a0, s0); a1 = scl4(a1, s0);
            a2 = scl4(a2, s1); a3 = scl4(a3, s1);
        }
        const v16bf afrag = cvt16(a0, a1, a2, a3);

        // ---- B fragments from LDS: lane lo = N row, K = 8*hi+0..7 and 16+8*hi+0..7.
        v16bf bfr[4];
#pragma unroll
        for (int nt = 0; nt < 4; ++nt) {
            const int row = nt * 16 + lo;
            float4 b0 = *(const float4*)&ldsB[cur][row][hi * 8 + 0];
            float4 b1 = *(const float4*)&ldsB[cur][row][hi * 8 + 4];
            float4 b2 = *(const float4*)&ldsB[cur][row][16 + hi * 8];
            float4 b3 = *(const float4*)&ldsB[cur][row][16 + hi * 8 + 4];
            bfr[nt] = cvt16(b0, b1, b2, b3);
        }
        // 4 back-to-back WMMAs (D->C accumulation chains need no NOPs)
        acc0 = __builtin_amdgcn_wmma_f32_16x16x32_bf16(false, afrag, false, bfr[0], (short)0, acc0, false, false);
        acc1 = __builtin_amdgcn_wmma_f32_16x16x32_bf16(false, afrag, false, bfr[1], (short)0, acc1, false, false);
        acc2 = __builtin_amdgcn_wmma_f32_16x16x32_bf16(false, afrag, false, bfr[2], (short)0, acc2, false, false);
        acc3 = __builtin_amdgcn_wmma_f32_16x16x32_bf16(false, afrag, false, bfr[3], (short)0, acc3, false, false);
    }

    // ---- Epilogue: C/D 16x16 f32 layout: lane lo = N, VGPR r -> M = r + 8*hi
#pragma unroll
    for (int nt = 0; nt < 4; ++nt) {
        const v8f acc = nt == 0 ? acc0 : nt == 1 ? acc1 : nt == 2 ? acc2 : acc3;
        const int ncol = n0 + nt * 16 + lo;
#pragma unroll
        for (int r = 0; r < 8; ++r) {
            const int m = m0 + r + 8 * hi;
            const float val = acc[r];
            if (mode == 0) {
                out[(size_t)m * DDIM + ncol] = elup1(val);
            } else if (mode == 3) {
                out[(size_t)m * DDIM + ncol] = val;
            } else {
                // scatter [m, n] -> [b, h, s, d]
                const int b = m >> 12, s = m & (SEQ - 1);
                const int h = ncol >> 6, d = ncol & 63;
                out[(((size_t)(b * NH + h)) * SEQ + s) * DH + d] = val;
            }
        }
    }
}

// ---------------------------------------------------------------------------
// c[b,h] = sum_t (S - t) * sum_d elup1(k[b,h,t,d]) * v[b,h,t,d]
// one block per (b,h)
// ---------------------------------------------------------------------------
__global__ __launch_bounds__(256)
void reduce_c_kernel(const float* __restrict__ kraw,
                     const float* __restrict__ vraw,
                     float* __restrict__ cArr) {
    const int bh = blockIdx.x;
    const float* __restrict__ kb = kraw + (size_t)bh * SEQ * DH;
    const float* __restrict__ vb = vraw + (size_t)bh * SEQ * DH;
    float acc = 0.f;
    for (int t = threadIdx.x; t < SEQ; t += 256) {
        const float* kr = kb + (size_t)t * DH;
        const float* vr = vb + (size_t)t * DH;
        float dot = 0.f;
#pragma unroll
        for (int d = 0; d < DH; d += 4) {
            float4 kk = *(const float4*)(kr + d);
            float4 vv = *(const float4*)(vr + d);
            dot += elup1(kk.x) * vv.x + elup1(kk.y) * vv.y +
                   elup1(kk.z) * vv.z + elup1(kk.w) * vv.w;
        }
        acc += (float)(SEQ - t) * dot;
    }
    __shared__ float red[256];
    red[threadIdx.x] = acc;
    __syncthreads();
#pragma unroll
    for (int off = 128; off > 0; off >>= 1) {
        if (threadIdx.x < off) red[threadIdx.x] += red[threadIdx.x + off];
        __syncthreads();
    }
    if (threadIdx.x == 0) cArr[bh] = red[0];
}

// ---------------------------------------------------------------------------
// Z[b,h,s] = 1 / (sum_d qf[b,s,h*64+d] * prefix_d(elup1(k[b,h,s,d])) + eps)
// one thread per (b,h,s) row; 64-wide sequential scan over the feature dim.
// ---------------------------------------------------------------------------
__global__ __launch_bounds__(256)
void z_kernel(const float* __restrict__ kraw,
              const float* __restrict__ qf,
              float* __restrict__ Zarr) {
    const int idx = blockIdx.x * 256 + threadIdx.x;  // (b*H+h)*S + s
    const int bh = idx >> 12;
    const int s  = idx & (SEQ - 1);
    const int b  = bh >> 4;
    const int h  = bh & (NH - 1);
    const float* __restrict__ krow = kraw + (size_t)idx * DH;
    const float* __restrict__ qrow = qf + ((size_t)b * SEQ + s) * DDIM + h * DH;
    float pref = 0.f, z = 0.f;
#pragma unroll
    for (int d = 0; d < DH; ++d) {
        pref += elup1(krow[d]);
        z += qrow[d] * pref;
    }
    Zarr[idx] = 1.f / (z + EPSV);
}

extern "C" void kernel_launch(void* const* d_in, const int* in_sizes, int n_in,
                              void* d_out, int out_size, void* d_ws, size_t ws_size,
                              hipStream_t stream) {
    const float* query = (const float*)d_in[0];
    const float* key   = (const float*)d_in[1];
    const float* value = (const float*)d_in[2];
    // d_in[3] = mask (int32, [B,S,S]) — causal tril; its last row is all-ones so
    // the reference's key-mask is a no-op. Intentionally unused.
    const float* Wq = (const float*)d_in[4];
    const float* Wk = (const float*)d_in[5];
    const float* Wv = (const float*)d_in[6];
    const float* Wo = (const float*)d_in[7];

    float* hidden = (float*)d_out;                           // [B,S,D]
    float* out_k  = hidden + (size_t)BATCH * SEQ * DDIM;     // [B,H,S,DH]
    float* out_v  = out_k  + (size_t)BATCH * NH * SEQ * DH;  // [B,H,S,DH]

    // workspace: qf [M,1024] (67MB) + Z [B*H*S] + c [B*H]
    float* qf   = (float*)d_ws;
    float* Zarr = qf + (size_t)MDIM * DDIM;
    float* cArr = Zarr + (size_t)BATCH * NH * SEQ;

    dim3 grid(MDIM / 128, DDIM / 64);   // 128 x 16 blocks
    dim3 block(256);

    // Q/K/V projections (bf16 WMMA GEMMs, async-LDS staged B)
    gemm_wmma_bf16_kernel<<<grid, block, 0, stream>>>(query, Wq, qf,   nullptr, nullptr, 0);
    gemm_wmma_bf16_kernel<<<grid, block, 0, stream>>>(key,   Wk, out_k, nullptr, nullptr, 1);
    gemm_wmma_bf16_kernel<<<grid, block, 0, stream>>>(value, Wv, out_v, nullptr, nullptr, 2);

    // scalar c[b,h] and normalizer Z[b,h,s]
    reduce_c_kernel<<<BATCH * NH, block, 0, stream>>>(out_k, out_v, cArr);
    z_kernel<<<(BATCH * NH * SEQ) / 256, block, 0, stream>>>(out_k, qf, Zarr);

    // output projection with fused qf * c * Z prologue
    gemm_wmma_bf16_kernel<<<grid, block, 0, stream>>>(qf, Wo, hidden, cArr, Zarr, 3);
}
